// SparseAttention_32908039422022
// MI455X (gfx1250) — matrix operand order repaired and checked by reading, not compiled, not run on hardware
//
#include <hip/hip_runtime.h>

// ---------------------------------------------------------------------------
// Problem constants (match reference: B=2, T=2048, C=1024, H=16, HD=64, W=256)
// ---------------------------------------------------------------------------
#define T_SEQ  2048
#define NHEADS 16
#define HDIM   64
#define CDIM   1024
#define WIN    256

typedef __attribute__((ext_vector_type(16))) __bf16 v16bf;
typedef __attribute__((ext_vector_type(8)))  __bf16 v8bf;
typedef __attribute__((ext_vector_type(8)))  float  v8f;
typedef __attribute__((ext_vector_type(4)))  float  v4f;

__device__ __forceinline__ __bf16 to_bf16(float f) { return (__bf16)f; }

// pack two f32 -> two bf16 in one dword (lo = a, hi = b)
__device__ __forceinline__ unsigned pack_bf16x2(float a, float b) {
  unsigned short la = __builtin_bit_cast(unsigned short, (__bf16)a);
  unsigned short hb = __builtin_bit_cast(unsigned short, (__bf16)b);
  return (unsigned)la | ((unsigned)hb << 16);
}

__device__ __forceinline__ v16bf make_v16(v8bf lo, v8bf hi) {
  v16bf r;
#pragma unroll
  for (int i = 0; i < 8; i++) { r[i] = lo[i]; r[i + 8] = hi[i]; }
  return r;
}

__device__ __forceinline__ v8f v8f_zero() {
  v8f r;
#pragma unroll
  for (int i = 0; i < 8; i++) r[i] = 0.0f;
  return r;
}

__device__ __forceinline__ v8f wmma_bf16(v16bf a, v16bf b, v8f c) {
  return __builtin_amdgcn_wmma_f32_16x16x32_bf16(false, a, false, b,
                                                 (short)0, c, false, false);
}

// ---------------------------------------------------------------------------
// Kernel 1/3: generic f32-in, bf16-WMMA GEMM.  C = A[M,K] * B[K,N] + bias
// mode 0: write f32 row-major into Cout
// mode 1: scatter qkv columns into bf16 q/k/v [B,H,T,HD] workspaces
// Block tile 128x128, K-step 32, 256 threads (8 waves), wave tile 32x64.
// ---------------------------------------------------------------------------
#define BM 128
#define BN 128
#define BK 32

__global__ __launch_bounds__(256)
void gemm_bf16_kernel(const float* __restrict__ A, const float* __restrict__ Bm,
                      const float* __restrict__ bias, float* __restrict__ Cout,
                      int M, int N, int K, int mode,
                      __bf16* __restrict__ qb, __bf16* __restrict__ kb,
                      __bf16* __restrict__ vb) {
  __shared__ __bf16 ldsA[BM * BK];   // [m][k]
  __shared__ __bf16 ldsB[BN * BK];   // transposed: [n][k]

  const int tid  = threadIdx.x;
  const int lane = tid & 31;
  const int wid  = tid >> 5;
  const int wm   = wid & 3;          // wave row  (0..3) -> 32 rows each
  const int wn   = wid >> 2;         // wave col  (0..1) -> 64 cols each
  const int lh   = lane >> 4;        // half-wave index (0/1)
  const int ll   = lane & 15;
  const int m0   = blockIdx.y * BM;
  const int n0   = blockIdx.x * BN;

  v8f acc[2][4];
#pragma unroll
  for (int i = 0; i < 2; i++)
#pragma unroll
    for (int j = 0; j < 4; j++) acc[i][j] = v8f_zero();

  // A staging: thread -> (row, 16-wide k segment); packed dword stores
  const int arow  = tid >> 1;          // 0..127
  const int akseg = (tid & 1) * 16;    // 0/16
  // B staging: thread -> (even k-row pair, 8-wide n segment); packed transposed
  const int bkk   = (tid & 15) * 2;    // 0,2,..,30
  const int bnseg = (tid >> 4) * 8;    // 0..120

  for (int k0 = 0; k0 < K; k0 += BK) {
    // ---- stage A tile (f32 -> bf16, row-major [128][32]) ----
    {
      const float* aptr = A + (size_t)(m0 + arow) * K + k0 + akseg;
      float tmp[16];
#pragma unroll
      for (int i = 0; i < 16; i += 4) {
        v4f v = *(const v4f*)(aptr + i);
        tmp[i + 0] = v.x; tmp[i + 1] = v.y; tmp[i + 2] = v.z; tmp[i + 3] = v.w;
      }
#pragma unroll
      for (int i = 0; i < 16; i += 2)
        *(unsigned*)&ldsA[arow * BK + akseg + i] = pack_bf16x2(tmp[i], tmp[i + 1]);
      if (k0 + BK < K) __builtin_prefetch(aptr + BK, 0, 0);
    }
    // ---- stage B tile transposed (f32 [32][128] -> bf16 [128][32]) ----
    {
      const float* b0 = Bm + (size_t)(k0 + bkk) * N + n0 + bnseg;
      const float* b1 = b0 + N;
      float r0[8], r1[8];
#pragma unroll
      for (int i = 0; i < 8; i += 4) {
        v4f x = *(const v4f*)(b0 + i);
        v4f y = *(const v4f*)(b1 + i);
        r0[i + 0] = x.x; r0[i + 1] = x.y; r0[i + 2] = x.z; r0[i + 3] = x.w;
        r1[i + 0] = y.x; r1[i + 1] = y.y; r1[i + 2] = y.z; r1[i + 3] = y.w;
      }
#pragma unroll
      for (int i = 0; i < 8; i++)
        *(unsigned*)&ldsB[(bnseg + i) * BK + bkk] = pack_bf16x2(r0[i], r1[i]);
      if (k0 + BK < K) __builtin_prefetch(b0 + (size_t)BK * N, 0, 0);
    }
    __syncthreads();

    // ---- load fragments (same contiguous k = lh*16 + e order for A and B) ----
    v16bf af[2], bfm[4];
#pragma unroll
    for (int mt = 0; mt < 2; mt++) {
      // A 16x32: lane ll = row; elems e at K = lh*16 + e
      const __bf16* p = &ldsA[(wm * 32 + mt * 16 + ll) * BK + lh * 16];
      af[mt] = make_v16(*(const v8bf*)p, *(const v8bf*)(p + 8));
    }
#pragma unroll
    for (int nt = 0; nt < 4; nt++) {
      // B 32x16: lane ll = col; elems e at K = lh*16 + e (contiguous in ldsB row)
      const __bf16* p = &ldsB[(wn * 64 + nt * 16 + ll) * BK + lh * 16];
      bfm[nt] = make_v16(*(const v8bf*)p, *(const v8bf*)(p + 8));
    }
#pragma unroll
    for (int mt = 0; mt < 2; mt++)
#pragma unroll
      for (int nt = 0; nt < 4; nt++)
        acc[mt][nt] = wmma_bf16(af[mt], bfm[nt], acc[mt][nt]);
    __syncthreads();
  }

  // ---- epilogue: C/D layout  M = r + 8*lh, N = ll ----
#pragma unroll
  for (int mt = 0; mt < 2; mt++) {
#pragma unroll
    for (int nt = 0; nt < 4; nt++) {
#pragma unroll
      for (int r = 0; r < 8; r++) {
        int grow = m0 + wm * 32 + mt * 16 + r + lh * 8;
        int gcol = n0 + wn * 64 + nt * 16 + ll;
        float val = acc[mt][nt][r] + bias[gcol];
        if (mode == 0) {
          Cout[(size_t)grow * N + gcol] = val;
        } else {
          int bb = grow / T_SEQ;
          int tt = grow - bb * T_SEQ;
          int which = gcol >> 10;          // 0=q, 1=k, 2=v
          int hh = (gcol >> 6) & (NHEADS - 1);
          int dd = gcol & (HDIM - 1);
          __bf16* dst = (which == 0) ? qb : ((which == 1) ? kb : vb);
          dst[(((size_t)bb * NHEADS + hh) * T_SEQ + tt) * HDIM + dd] = to_bf16(val);
        }
      }
    }
  }
}

// ---------------------------------------------------------------------------
// Kernel 2: sliding-window causal attention, flash-style online softmax.
// Block = (b, h, 64 q-rows); 128 threads = 4 waves, 16 q-rows per wave.
// Key blocks of 32; QK^T and P*V via v_wmma_f32_16x16x32_bf16.
// ---------------------------------------------------------------------------
__global__ __launch_bounds__(128)
void attn_kernel(const __bf16* __restrict__ qb, const __bf16* __restrict__ kb,
                 const __bf16* __restrict__ vb, float* __restrict__ y) {
  __shared__ __bf16 Vt[HDIM * 32];     // V transposed: [hd][key]
  __shared__ __bf16 Pw[4][16][32];     // per-wave P tile (C-layout -> A-layout)

  const int tid  = threadIdx.x;
  const int lane = tid & 31;
  const int wt   = tid >> 5;
  const int lh   = lane >> 4;
  const int ll   = lane & 15;
  const int b    = blockIdx.z;
  const int h    = blockIdx.y;
  const int q0   = blockIdx.x * 64;
  const int bh   = b * NHEADS + h;
  const int qw   = q0 + wt * 16;

  // Q fragments (A-layout), two K-steps over HD=64
  v16bf qf[2];
  {
    const __bf16* qrow = qb + ((size_t)bh * T_SEQ + (qw + ll)) * HDIM;
#pragma unroll
    for (int st = 0; st < 2; st++) {
      const __bf16* p = qrow + st * 32 + lh * 16;
      qf[st] = make_v16(*(const v8bf*)p, *(const v8bf*)(p + 8));
    }
  }

  v8f acc[4];
#pragma unroll
  for (int ht = 0; ht < 4; ht++) acc[ht] = v8f_zero();
  float mrow[8], lrow[8];
#pragma unroll
  for (int r = 0; r < 8; r++) { mrow[r] = -3.0e38f; lrow[r] = 0.0f; }

  int jlo = q0 - WIN;
  if (jlo < 0) jlo = 0;
  const int jstart = jlo & ~31;

  for (int j0 = jstart; j0 <= q0 + 63; j0 += 32) {
    // ---- cooperatively stage V[j0:j0+32][0:64] transposed into LDS ----
    {
      const int kk  = tid & 31;
      const int seg = (tid >> 5) * 16;
      const __bf16* vrow = vb + ((size_t)bh * T_SEQ + (j0 + kk)) * HDIM + seg;
      v8bf v0 = *(const v8bf*)vrow;
      v8bf v1 = *(const v8bf*)(vrow + 8);
#pragma unroll
      for (int i = 0; i < 8; i++) {
        Vt[(seg + i)     * 32 + kk] = v0[i];
        Vt[(seg + 8 + i) * 32 + kk] = v1[i];
      }
    }
    __syncthreads();

    // block-level window cull (wave-uniform)
    bool active = (j0 <= qw + 15) && (j0 + 31 >= qw - WIN);
    if (active) {
      // ---- S = Q K^T  (two 16x16 key tiles) ----
      v8f sc[2];
#pragma unroll
      for (int nt = 0; nt < 2; nt++) {
        v8f s = v8f_zero();
#pragma unroll
        for (int st = 0; st < 2; st++) {
          // B-layout column of K^T == contiguous row of K
          const __bf16* krow =
              kb + ((size_t)bh * T_SEQ + (j0 + nt * 16 + ll)) * HDIM + st * 32 + lh * 16;
          v16bf kf = make_v16(*(const v8bf*)krow, *(const v8bf*)(krow + 8));
          s = wmma_bf16(qf[st], kf, s);
        }
        // mask (causal + window) and scale by 1/sqrt(64)
#pragma unroll
        for (int r = 0; r < 8; r++) {
          int row = qw + r + lh * 8;
          int key = j0 + nt * 16 + ll;
          bool ok = (key <= row) && (row - key <= WIN);
          s[r] = ok ? s[r] * 0.125f : -3.0e38f;
        }
        sc[nt] = s;
      }

      // ---- online softmax (row stats replicated across each 16-lane half) ----
#pragma unroll
      for (int r = 0; r < 8; r++) {
        float tm = fmaxf(sc[0][r], sc[1][r]);
#pragma unroll
        for (int mk = 1; mk < 16; mk <<= 1) tm = fmaxf(tm, __shfl_xor(tm, mk, 32));
        float mn    = fmaxf(mrow[r], tm);
        float alpha = __expf(mrow[r] - mn);
        float e0    = __expf(sc[0][r] - mn);
        float e1    = __expf(sc[1][r] - mn);
        float rs    = e0 + e1;
#pragma unroll
        for (int mk = 1; mk < 16; mk <<= 1) rs += __shfl_xor(rs, mk, 32);
        lrow[r] = lrow[r] * alpha + rs;
        mrow[r] = mn;
#pragma unroll
        for (int ht = 0; ht < 4; ht++) acc[ht][r] *= alpha;
        // C-layout -> LDS (row = r + 8*lh, key = nt*16 + ll)
        Pw[wt][r + lh * 8][ll]      = to_bf16(e0);
        Pw[wt][r + lh * 8][16 + ll] = to_bf16(e1);
      }
      // intra-wave LDS store->load ordering
      asm volatile("s_wait_dscnt 0x0" ::: "memory");

      // P fragment (A-layout 16x32 over keys)
      const __bf16* pp = &Pw[wt][ll][lh * 16];
      v16bf pf = make_v16(*(const v8bf*)pp, *(const v8bf*)(pp + 8));

      // ---- O += P * V  (4 HD tiles) ----
#pragma unroll
      for (int ht = 0; ht < 4; ht++) {
        const __bf16* vp = &Vt[(ht * 16 + ll) * 32 + lh * 16];
        v16bf vf = make_v16(*(const v8bf*)vp, *(const v8bf*)(vp + 8));
        acc[ht] = wmma_bf16(pf, vf, acc[ht]);
      }
    }
    __syncthreads();
  }

  // ---- normalize and write y_attn as f32 [B,T,C] (input of proj GEMM) ----
#pragma unroll
  for (int ht = 0; ht < 4; ht++) {
#pragma unroll
    for (int r = 0; r < 8; r++) {
      int row = qw + r + lh * 8;
      int col = h * HDIM + ht * 16 + ll;
      y[((size_t)b * T_SEQ + row) * CDIM + col] = acc[ht][r] / lrow[r];
    }
  }
}

// ---------------------------------------------------------------------------
// Launch: QKV GEMM (scatter to bf16 q/k/v) -> attention -> proj GEMM
// Workspace: q/k/v bf16 (3 x 8 MB) + y_attn f32 (16 MB) = 40 MB
// ---------------------------------------------------------------------------
extern "C" void kernel_launch(void* const* d_in, const int* in_sizes, int n_in,
                              void* d_out, int out_size, void* d_ws, size_t ws_size,
                              hipStream_t stream) {
  const float* x     = (const float*)d_in[0];
  const float* Wqkv  = (const float*)d_in[1];
  const float* bqkv  = (const float*)d_in[2];
  const float* Wproj = (const float*)d_in[3];
  const float* bproj = (const float*)d_in[4];
  // d_in[5] (mask) unused: window mask computed analytically (matches reference)
  (void)in_sizes; (void)n_in; (void)out_size; (void)ws_size;

  const int Bsz = 2;
  const int M   = Bsz * T_SEQ;                              // 4096
  const size_t nqkv = (size_t)Bsz * NHEADS * T_SEQ * HDIM;  // 4,194,304

  __bf16* qb = (__bf16*)d_ws;
  __bf16* kb = qb + nqkv;
  __bf16* vb = kb + nqkv;
  float*  ya = (float*)(vb + nqkv);
  float*  out = (float*)d_out;

  dim3 blk(256);
  dim3 g1((3 * CDIM) / BN, M / BM);   // 24 x 32
  gemm_bf16_kernel<<<g1, blk, 0, stream>>>(x, Wqkv, bqkv, nullptr,
                                           M, 3 * CDIM, CDIM, 1, qb, kb, vb);

  dim3 g2(T_SEQ / 64, NHEADS, Bsz);   // 32 x 16 x 2
  attn_kernel<<<g2, dim3(128), 0, stream>>>(qb, kb, vb, ya);

  dim3 g3(CDIM / BN, M / BM);         // 8 x 32
  gemm_bf16_kernel<<<g3, blk, 0, stream>>>(ya, Wproj, bproj, out,
                                           M, CDIM, CDIM, 0, nullptr, nullptr, nullptr);
}
